// FlexibleCNN_88321707475735
// MI455X (gfx1250) — compile-verified, matrix operand order checked by reading, not run on hardware
//
#include <hip/hip_runtime.h>
#include <hip/hip_bf16.h>

typedef __attribute__((ext_vector_type(16))) _Float16 v16h;
typedef __attribute__((ext_vector_type(8)))  _Float16 v8h;
typedef __attribute__((ext_vector_type(8)))  float    v8f;

#define BATCH 32
#define EPS 1e-5f

// Fragment-order index for B operand of v_wmma_f32_16x16x32_f16.
// Lane (h*16+lrow) holds col o=nt*16+lrow; its 16 halfs are
// K = {h*8 .. h*8+7} U {16+h*8 .. 16+h*8+7} of the 32-K chunk, contiguous.
__device__ __host__ inline int frag_index(int k, int o, int NT) {
    int kc = k >> 5, kk = k & 31;
    int grp = kk >> 4, rem = kk & 15;
    int h = rem >> 3, e = grp * 8 + (rem & 7);
    int nt = o >> 4, lrow = o & 15;
    int lane = h * 16 + lrow;
    return ((kc * NT + nt) * 32 + lane) * 16 + e;
}

// ---------------------------------------------------------------------------
// Kernel 0: weight-gen MLP (ind==ones -> identical weights for all samples),
// BN folding, and storage in WMMA-fragment order (f16).
// ---------------------------------------------------------------------------
__global__ __launch_bounds__(256) void prep_kernel(
    const float* __restrict__ w1, const float* __restrict__ b1,
    const float* __restrict__ w2, const float* __restrict__ b2,
    const float* __restrict__ w3, const float* __restrict__ b3,
    const float* __restrict__ wb, const float* __restrict__ bb,
    const float* __restrict__ c2w, const float* __restrict__ c2b,
    const float* __restrict__ c3w, const float* __restrict__ c3b,
    const float* __restrict__ g1, const float* __restrict__ be1,
    const float* __restrict__ m1, const float* __restrict__ v1,
    const float* __restrict__ g2, const float* __restrict__ be2,
    const float* __restrict__ m2, const float* __restrict__ v2,
    const float* __restrict__ g3, const float* __restrict__ be3,
    const float* __restrict__ m3, const float* __restrict__ v3,
    _Float16* __restrict__ w1f, float* __restrict__ b1f,
    _Float16* __restrict__ w2f, float* __restrict__ b2f,
    _Float16* __restrict__ w3f, float* __restrict__ b3f)
{
    __shared__ float h1[128];
    __shared__ float h2[256];
    int t = threadIdx.x;

    if (t < 128) {
        float s = b1[t];
        for (int c = 0; c < 3; ++c) s += w1[t * 3 + c];
        h1[t] = fmaxf(s, 0.f);
    }
    __syncthreads();
    if (t < 256) {
        float s = b2[t];
        for (int k = 0; k < 128; ++k) s += w2[t * 128 + k] * h1[k];
        h2[t] = fmaxf(s, 0.f);
    }
    __syncthreads();

    // conv1 dynamic weights, BN1-folded, K padded 27->32, NT=2
    for (int i = t; i < 32 * 32; i += 256) {
        int k = i >> 5, o = i & 31;
        float val = 0.f;
        if (k < 27) {
            int widx = o * 27 + k;
            float s = b3[widx];
            for (int q = 0; q < 256; ++q) s += w3[widx * 256 + q] * h2[q];
            val = s;
        }
        float inv = g1[o] * rsqrtf(v1[o] + EPS);
        w1f[frag_index(k, o, 2)] = (_Float16)(val * inv);
    }
    if (t < 32) {
        float db = bb[t];
        for (int c = 0; c < 3; ++c) db += wb[t * 3 + c];
        float inv = g1[t] * rsqrtf(v1[t] + EPS);
        b1f[t] = (db - m1[t]) * inv + be1[t];
    }

    // conv2 weights BN2-folded, K=288, NT=4
    for (int i = t; i < 288 * 64; i += 256) {
        int k = i >> 6, o = i & 63;
        float inv = g2[o] * rsqrtf(v2[o] + EPS);
        w2f[frag_index(k, o, 4)] = (_Float16)(c2w[o * 288 + k] * inv);
    }
    if (t < 64) {
        float inv = g2[t] * rsqrtf(v2[t] + EPS);
        b2f[t] = (c2b[t] - m2[t]) * inv + be2[t];
    }

    // conv3 weights BN3-folded, K=576, NT=8
    for (int i = t; i < 576 * 128; i += 256) {
        int k = i >> 7, o = i & 127;
        float inv = g3[o] * rsqrtf(v3[o] + EPS);
        w3f[frag_index(k, o, 8)] = (_Float16)(c3w[o * 576 + k] * inv);
    }
    if (t < 128) {
        float inv = g3[t] * rsqrtf(v3[t] + EPS);
        b3f[t] = (c3b[t] - m3[t]) * inv + be3[t];
    }
}

// ---------------------------------------------------------------------------
// Kernel 1: fp32 -> fp16
// ---------------------------------------------------------------------------
__global__ __launch_bounds__(256) void f32_to_f16(const float* __restrict__ in,
                                                  _Float16* __restrict__ out, int n)
{
    for (int i = blockIdx.x * 256 + threadIdx.x; i < n; i += gridDim.x * 256)
        out[i] = (_Float16)in[i];
}

// ---------------------------------------------------------------------------
// Kernel 2: 3x3 pad-1 conv, implicit GEMM on v_wmma_f32_16x16x32_f16.
// Double-buffered LDS, B pre-permuted to fragment order, B128 fragment loads.
// Block = 256 threads (8 wave32); tile = 2 rows x 32 cols x COUT.
// ---------------------------------------------------------------------------
template <int CIN, int COUT, int HW, bool POOL>
__global__ __launch_bounds__(256) void conv3x3_wmma(
    const _Float16* __restrict__ in,   // [N][CIN][HW][HW]
    const _Float16* __restrict__ wfr,  // fragment-ordered folded weights
    const float* __restrict__ bias,    // [COUT] fused bias
    _Float16* __restrict__ out16,      // pooled output (POOL)
    float* __restrict__ out32)         // full-res output (!POOL)
{
    constexpr int KTOT = CIN * 9;
    constexpr int NK = (KTOT + 31) / 32;
    constexpr int NT = COUT / 16;       // N tiles
    constexpr int TPW = (4 * NT) / 8;   // WMMA tiles per wave (1,2,4); TPW|NT
    constexpr int TILES_C = (HW + 31) / 32;
    constexpr int BCH = 32 * COUT;      // halfs per B chunk

    int gid = blockIdx.x;
    int tc = gid % TILES_C; gid /= TILES_C;
    int tr = gid % (HW / 2); gid /= (HW / 2);
    int n = gid;
    int oh0 = tr * 2, ow0 = tc * 32;

    __shared__ __align__(16) _Float16 As[2][64 * 32];
    __shared__ __align__(16) _Float16 Bs[2][BCH];
    __shared__ float Ps[64 * COUT];

    int tid = threadIdx.x;
    int lane = tid & 31, wave = tid >> 5;
    int halfsel = lane >> 4, lrow = lane & 15;

    // every wave owns one M-tile (TPW divides NT)
    int mt = (wave * TPW) / NT;
    int nt0 = (wave * TPW) % NT;

    v8f acc[TPW] = {};

    const _Float16* inN = in + (size_t)n * CIN * HW * HW;

    auto stageA = [&](int kc, _Float16* dst) {
        for (int idx = tid; idx < 64 * 32; idx += 256) {
            int m = idx >> 5, kk = idx & 31;
            int k = kc * 32 + kk;
            _Float16 val = (_Float16)0.f;
            int lc = m & 31, lr = m >> 5;
            int ow = ow0 + lc, oh = oh0 + lr;
            if (k < KTOT && ow < HW) {
                int ci = k / 9, r = k - ci * 9;
                int kh = r / 3, kw = r - kh * 3;
                int ih = oh + kh - 1, iw = ow + kw - 1;
                if (ih >= 0 && ih < HW && iw >= 0 && iw < HW)
                    val = inN[((size_t)ci * HW + ih) * HW + iw];
            }
            dst[idx] = val;
        }
    };
    auto stageB = [&](int kc, _Float16* dst) {
        const v8h* src = (const v8h*)(wfr + (size_t)kc * BCH);
        v8h* d = (v8h*)dst;
        for (int idx = tid; idx < BCH / 8; idx += 256) d[idx] = src[idx];
        if (kc + 1 < NK)
            __builtin_prefetch(wfr + (size_t)(kc + 1) * BCH, 0, 0);
    };

    stageA(0, As[0]);
    stageB(0, Bs[0]);
    __syncthreads();

    for (int kc = 0; kc < NK; ++kc) {
        int cur = kc & 1;
        if (kc + 1 < NK) {
            stageA(kc + 1, As[cur ^ 1]);
            stageB(kc + 1, Bs[cur ^ 1]);
        }
        // A fragment: two contiguous 8-half (128-bit) LDS loads
        const _Float16* arow = &As[cur][(mt * 16 + lrow) * 32 + halfsel * 8];
        v8h alo = *(const v8h*)(arow);
        v8h ahi = *(const v8h*)(arow + 16);
        v16h a = __builtin_shufflevector(alo, ahi, 0, 1, 2, 3, 4, 5, 6, 7,
                                         8, 9, 10, 11, 12, 13, 14, 15);
#pragma unroll
        for (int t = 0; t < TPW; ++t) {
            // B fragment: 16 contiguous halfs (fragment-ordered in LDS)
            const _Float16* bp = &Bs[cur][(((nt0 + t) * 32) + lane) * 16];
            v8h blo = *(const v8h*)(bp);
            v8h bhi = *(const v8h*)(bp + 8);
            v16h b = __builtin_shufflevector(blo, bhi, 0, 1, 2, 3, 4, 5, 6, 7,
                                             8, 9, 10, 11, 12, 13, 14, 15);
            acc[t] = __builtin_amdgcn_wmma_f32_16x16x32_f16(
                false, a, false, b, (short)0, acc[t], false, false);
        }
        __syncthreads();
    }

    // epilogue: bias + ReLU into LDS transpose buffer
#pragma unroll
    for (int t = 0; t < TPW; ++t) {
        int o = (nt0 + t) * 16 + lrow;
        float bo = bias[o];
#pragma unroll
        for (int v = 0; v < 8; ++v) {
            int m = mt * 16 + v + 8 * halfsel; // C/D VGPR layout
            Ps[o * 64 + m] = fmaxf(acc[t][v] + bo, 0.f);
        }
    }
    __syncthreads();

    if constexpr (POOL) {
        const int HWo = HW / 2;
        for (int idx = tid; idx < COUT * 16; idx += 256) {
            int o = idx >> 4, j = idx & 15;
            if (ow0 + 2 * j < HW) {
                float v00 = Ps[o * 64 + 2 * j];
                float v01 = Ps[o * 64 + 2 * j + 1];
                float v10 = Ps[o * 64 + 32 + 2 * j];
                float v11 = Ps[o * 64 + 32 + 2 * j + 1];
                float mx = fmaxf(fmaxf(v00, v01), fmaxf(v10, v11));
                out16[(((size_t)n * COUT + o) * HWo + tr) * HWo + (ow0 >> 1) + j] =
                    (_Float16)mx;
            }
        }
    } else {
        // coalesced: consecutive threads -> consecutive ow
        for (int idx = tid; idx < COUT * 64; idx += 256) {
            int o = idx >> 6, m = idx & 63;
            int lr = m >> 5, lc = m & 31;
            int oh = oh0 + lr, ow = ow0 + lc;
            if (ow < HW)
                out32[(((size_t)n * COUT + o) * HW + oh) * HW + ow] =
                    Ps[o * 64 + m];
        }
    }
}

// ---------------------------------------------------------------------------
// Kernel 3: global average pool, one block per (n, channel)
// ---------------------------------------------------------------------------
__global__ __launch_bounds__(256) void avgpool_kernel(const float* __restrict__ c3,
                                                      float* __restrict__ feat)
{
    int nc = blockIdx.x;
    const float* p = c3 + (size_t)nc * (56 * 56);
    float s = 0.f;
    for (int i = threadIdx.x; i < 56 * 56; i += 256) s += p[i];
    __shared__ float red[256];
    red[threadIdx.x] = s;
    __syncthreads();
    for (int off = 128; off; off >>= 1) {
        if (threadIdx.x < off) red[threadIdx.x] += red[threadIdx.x + off];
        __syncthreads();
    }
    if (threadIdx.x == 0) feat[nc] = red[0] * (1.0f / (56.0f * 56.0f));
}

// ---------------------------------------------------------------------------
// Kernel 4: FC 128 -> 100 (fp32, tiny)
// ---------------------------------------------------------------------------
__global__ __launch_bounds__(256) void fc_kernel(const float* __restrict__ feat,
                                                 const float* __restrict__ fcw,
                                                 const float* __restrict__ fcb,
                                                 float* __restrict__ out)
{
    int i = blockIdx.x * 256 + threadIdx.x;
    if (i >= BATCH * 100) return;
    int n = i / 100, c = i - n * 100;
    float s = fcb[c];
    for (int k = 0; k < 128; ++k) s += feat[n * 128 + k] * fcw[c * 128 + k];
    out[i] = s;
}

// ---------------------------------------------------------------------------
extern "C" void kernel_launch(void* const* d_in, const int* in_sizes, int n_in,
                              void* d_out, int out_size, void* d_ws, size_t ws_size,
                              hipStream_t stream)
{
    (void)in_sizes; (void)n_in; (void)out_size; (void)ws_size;
    const float* x   = (const float*)d_in[0];
    const float* w1  = (const float*)d_in[1];
    const float* b1  = (const float*)d_in[2];
    const float* w2  = (const float*)d_in[3];
    const float* b2  = (const float*)d_in[4];
    const float* w3  = (const float*)d_in[5];
    const float* b3  = (const float*)d_in[6];
    const float* wb  = (const float*)d_in[7];
    const float* bb  = (const float*)d_in[8];
    const float* c2w = (const float*)d_in[9];
    const float* c2b = (const float*)d_in[10];
    const float* c3w = (const float*)d_in[11];
    const float* c3b = (const float*)d_in[12];
    const float* g1  = (const float*)d_in[13];
    const float* be1 = (const float*)d_in[14];
    const float* m1  = (const float*)d_in[15];
    const float* v1  = (const float*)d_in[16];
    const float* g2  = (const float*)d_in[17];
    const float* be2 = (const float*)d_in[18];
    const float* m2  = (const float*)d_in[19];
    const float* v2  = (const float*)d_in[20];
    const float* g3  = (const float*)d_in[21];
    const float* be3 = (const float*)d_in[22];
    const float* m3  = (const float*)d_in[23];
    const float* v3  = (const float*)d_in[24];
    const float* fcw = (const float*)d_in[25];
    const float* fcb = (const float*)d_in[26];
    float* out = (float*)d_out;

    char* ws = (char*)d_ws;
    size_t off = 0;
    auto carve = [&](size_t bytes) -> char* {
        char* p = ws + off;
        off = (off + bytes + 255) & ~(size_t)255;
        return p;
    };
    _Float16* w1f = (_Float16*)carve(32 * 32 * 2);
    _Float16* w2f = (_Float16*)carve(288 * 64 * 2);
    _Float16* w3f = (_Float16*)carve(576 * 128 * 2);
    float*    b1f = (float*)carve(32 * 4);
    float*    b2f = (float*)carve(64 * 4);
    float*    b3f = (float*)carve(128 * 4);
    _Float16* x16 = (_Float16*)carve((size_t)BATCH * 3 * 224 * 224 * 2);
    _Float16* p1  = (_Float16*)carve((size_t)BATCH * 32 * 112 * 112 * 2);
    _Float16* p2  = (_Float16*)carve((size_t)BATCH * 64 * 56 * 56 * 2);
    float*    c3o = (float*)carve((size_t)BATCH * 128 * 56 * 56 * 4);
    float*    feat = (float*)carve((size_t)BATCH * 128 * 4);

    prep_kernel<<<1, 256, 0, stream>>>(w1, b1, w2, b2, w3, b3, wb, bb,
                                       c2w, c2b, c3w, c3b,
                                       g1, be1, m1, v1, g2, be2, m2, v2,
                                       g3, be3, m3, v3,
                                       w1f, b1f, w2f, b2f, w3f, b3f);
    f32_to_f16<<<2048, 256, 0, stream>>>(x, x16, BATCH * 3 * 224 * 224);
    conv3x3_wmma<3, 32, 224, true>
        <<<BATCH * 112 * 7, 256, 0, stream>>>(x16, w1f, b1f, p1, nullptr);
    conv3x3_wmma<32, 64, 112, true>
        <<<BATCH * 56 * 4, 256, 0, stream>>>(p1, w2f, b2f, p2, nullptr);
    conv3x3_wmma<64, 128, 56, false>
        <<<BATCH * 28 * 2, 256, 0, stream>>>(p2, w3f, b3f, nullptr, c3o);
    avgpool_kernel<<<BATCH * 128, 256, 0, stream>>>(c3o, feat);
    fc_kernel<<<(BATCH * 100 + 255) / 256, 256, 0, stream>>>(feat, fcw, fcb, out);
}
